// LRU_58557584113818
// MI455X (gfx1250) — compile-verified, matrix operand order checked by reading, not run on hardware
//
#include <hip/hip_runtime.h>

// Problem constants (match reference)
#define BB   16
#define LL   4096
#define FF   512
#define HH   512
#define MTOT (BB * LL)   // 65536 rows of the GEMM
#define CT   256         // scan chunk length
#define NC   (LL / CT)   // 16 chunks per sequence
#define KSTEPS (FF / 64) // 8 K-panels of width 64

typedef __attribute__((ext_vector_type(16))) __bf16 v16bf;
typedef __attribute__((ext_vector_type(8)))  __bf16 v8bf;
typedef __attribute__((ext_vector_type(8)))  float  v8f;

union BF16x16 { v16bf v; v8bf h[2]; };

__device__ __forceinline__ __bf16 f2bf(float f) {
  // round-to-nearest-even float -> bf16
  unsigned u = __float_as_uint(f);
  u = (u + 0x7FFFu + ((u >> 16) & 1u)) >> 16;
  unsigned short s = (unsigned short)u;
  return __builtin_bit_cast(__bf16, s);
}

// ---------------------------------------------------------------------------
// Kernel 1: dual GEMM  BxR = (x @ B_real) * delta, BxI = (x @ B_img) * delta
// Block: 256 threads = 8 wave32s; tile 128(M) x 64(N); K stepped by 64.
// Double-buffered LDS: next panel's global loads are issued before the WMMA
// block, converted+stored to the alternate buffer after, 1 barrier / K-step.
// ---------------------------------------------------------------------------
__global__ __launch_bounds__(256)
void lru_gemm_bf16(const float* __restrict__ x,
                   const float* __restrict__ Breal,
                   const float* __restrict__ Bimg,
                   const float* __restrict__ delta,
                   float* __restrict__ BxR,
                   float* __restrict__ BxI) {
  __shared__ __bf16 As[2][128][72];   // A tiles (pad 72 -> 144B rows, 16B aligned)
  __shared__ __bf16 BrT[2][64][72];   // B_real tiles, TRANSPOSED: [n][k]
  __shared__ __bf16 BiT[2][64][72];   // B_img  tiles, TRANSPOSED: [n][k]

  const int tid  = threadIdx.x;
  const int lane = tid & 31;
  const int wave = tid >> 5;
  const int wm   = wave & 3;       // wave's M slot (0..3) -> 32 rows each
  const int wn   = wave >> 2;      // wave's N slot (0..1) -> 32 cols each
  const int lr   = lane & 15;      // row/col within 16x16 tile
  const int lh   = lane >> 4;      // K-half / M-half selector

  const int nBlk = blockIdx.x * 64;
  const int mBlk = blockIdx.y * 128;

  const int fquad = tid & 15;      // float4 slot along a 64-wide row
  const int row0  = tid >> 4;      // 0..15

  v8f acc[2][2][2];                // [mt][nt][real/imag]
#pragma unroll
  for (int a = 0; a < 2; ++a)
#pragma unroll
    for (int b = 0; b < 2; ++b)
#pragma unroll
      for (int c = 0; c < 2; ++c)
        acc[a][b][c] = (v8f){};

  float4 aReg[8], brReg[4], biReg[4];

  auto loadGlobal = [&](int k0) {
#pragma unroll
    for (int i = 0; i < 8; ++i)
      aReg[i] = *(const float4*)(x + (size_t)(mBlk + row0 + i * 16) * FF + k0 + fquad * 4);
#pragma unroll
    for (int i = 0; i < 4; ++i) {
      const size_t g = (size_t)(k0 + row0 + i * 16) * HH + nBlk + fquad * 4;
      brReg[i] = *(const float4*)(Breal + g);
      biReg[i] = *(const float4*)(Bimg  + g);
    }
  };

  auto storeLds = [&](int buf) {
#pragma unroll
    for (int i = 0; i < 8; ++i) {
      const int r = row0 + i * 16;
      As[buf][r][fquad * 4 + 0] = f2bf(aReg[i].x);
      As[buf][r][fquad * 4 + 1] = f2bf(aReg[i].y);
      As[buf][r][fquad * 4 + 2] = f2bf(aReg[i].z);
      As[buf][r][fquad * 4 + 3] = f2bf(aReg[i].w);
    }
#pragma unroll
    for (int i = 0; i < 4; ++i) {
      const int k = row0 + i * 16;
      BrT[buf][fquad * 4 + 0][k] = f2bf(brReg[i].x);
      BrT[buf][fquad * 4 + 1][k] = f2bf(brReg[i].y);
      BrT[buf][fquad * 4 + 2][k] = f2bf(brReg[i].z);
      BrT[buf][fquad * 4 + 3][k] = f2bf(brReg[i].w);
      BiT[buf][fquad * 4 + 0][k] = f2bf(biReg[i].x);
      BiT[buf][fquad * 4 + 1][k] = f2bf(biReg[i].y);
      BiT[buf][fquad * 4 + 2][k] = f2bf(biReg[i].z);
      BiT[buf][fquad * 4 + 3][k] = f2bf(biReg[i].w);
    }
  };

  // prologue: stage panel 0
  loadGlobal(0);
  storeLds(0);
  __syncthreads();

  for (int kb = 0; kb < KSTEPS; ++kb) {
    const int cur = kb & 1;

    // issue next panel's global loads before compute (overlap with WMMA)
    if (kb + 1 < KSTEPS) loadGlobal((kb + 1) * 64);

    // ---- compute: two K=32 sub-steps on the staged 64-wide panel ----
#pragma unroll
    for (int ks = 0; ks < 64; ks += 32) {
      const int c1 = ks + lh * 8;        // K chunk {0..7} or {8..15}
      const int c2 = ks + 16 + lh * 8;   // K chunk {16..23} or {24..31}

      BF16x16 af[2], brf[2], bif[2];
#pragma unroll
      for (int mt = 0; mt < 2; ++mt) {
        const int r = wm * 32 + mt * 16 + lr;
        af[mt].h[0] = *(const v8bf*)&As[cur][r][c1];
        af[mt].h[1] = *(const v8bf*)&As[cur][r][c2];
      }
#pragma unroll
      for (int nt = 0; nt < 2; ++nt) {
        const int n = wn * 32 + nt * 16 + lr;
        brf[nt].h[0] = *(const v8bf*)&BrT[cur][n][c1];
        brf[nt].h[1] = *(const v8bf*)&BrT[cur][n][c2];
        bif[nt].h[0] = *(const v8bf*)&BiT[cur][n][c1];
        bif[nt].h[1] = *(const v8bf*)&BiT[cur][n][c2];
      }
#pragma unroll
      for (int mt = 0; mt < 2; ++mt)
#pragma unroll
        for (int nt = 0; nt < 2; ++nt) {
          acc[mt][nt][0] = __builtin_amdgcn_wmma_f32_16x16x32_bf16(
              false, af[mt].v, false, brf[nt].v, (short)0, acc[mt][nt][0], false, false);
          acc[mt][nt][1] = __builtin_amdgcn_wmma_f32_16x16x32_bf16(
              false, af[mt].v, false, bif[nt].v, (short)0, acc[mt][nt][1], false, false);
        }
    }

    // stage next panel into the alternate buffer, single barrier per step
    if (kb + 1 < KSTEPS) storeLds(cur ^ 1);
    __syncthreads();
  }

  // ---- epilogue: scale by delta[col], store fp32 planes ----
  float dl[2];
#pragma unroll
  for (int nt = 0; nt < 2; ++nt)
    dl[nt] = delta[nBlk + wn * 32 + nt * 16 + lr];

#pragma unroll
  for (int mt = 0; mt < 2; ++mt) {
    const int rbase = mBlk + wm * 32 + mt * 16 + lh * 8;  // D layout: lanes 16-31 are M+8
#pragma unroll
    for (int nt = 0; nt < 2; ++nt) {
      const int col = nBlk + wn * 32 + nt * 16 + lr;
#pragma unroll
      for (int v = 0; v < 8; ++v) {
        const size_t o = (size_t)(rbase + v) * HH + col;
        BxR[o] = acc[mt][nt][0][v] * dl[nt];
        BxI[o] = acc[mt][nt][1][v] * dl[nt];
      }
    }
  }
}

// ---------------------------------------------------------------------------
// Kernel 2a: per-chunk zero-init scan final value S_c = sum lam^{T-1-j} Bx_j
// grid: (HH/256, NC, BB)
// ---------------------------------------------------------------------------
__global__ __launch_bounds__(256)
void lru_chunk_sum(const float* __restrict__ BxR, const float* __restrict__ BxI,
                   const float* __restrict__ nu, const float* __restrict__ theta,
                   float* __restrict__ SR, float* __restrict__ SI) {
  const int h = blockIdx.x * blockDim.x + threadIdx.x;
  const int c = blockIdx.y;
  const int b = blockIdx.z;
  const float mag = __expf(-__expf(nu[h]));
  float sn, cs; __sincosf(theta[h], &sn, &cs);
  const float lr = mag * cs, li = mag * sn;

  const size_t base = ((size_t)b * LL + (size_t)c * CT) * HH + h;
  const float* pr = BxR + base;
  const float* pi = BxI + base;
  float sr = 0.f, si = 0.f;
#pragma unroll 4
  for (int j = 0; j < CT; ++j) {
    const float xr = pr[(size_t)j * HH];
    const float xi = pi[(size_t)j * HH];
    const float nr = fmaf(lr, sr, fmaf(-li, si, xr));
    const float ni = fmaf(lr, si, fmaf( li, sr, xi));
    sr = nr; si = ni;
  }
  const size_t o = (size_t)(b * NC + c) * HH + h;
  SR[o] = sr; SI[o] = si;
}

// ---------------------------------------------------------------------------
// Kernel 2b: sequential combine over NC=16 chunks: carry[c] = lam^T*carry + S
// carry[0] = h0 (initial state). lam^T in closed form (|lam|<1, underflow->0).
// grid: (HH/256, BB)
// ---------------------------------------------------------------------------
__global__ __launch_bounds__(256)
void lru_chunk_combine(const float* __restrict__ SR, const float* __restrict__ SI,
                       const float* __restrict__ nu, const float* __restrict__ theta,
                       const float* __restrict__ h0r, const float* __restrict__ h0i,
                       float* __restrict__ CR, float* __restrict__ CI) {
  const int h = blockIdx.x * blockDim.x + threadIdx.x;
  const int b = blockIdx.y;
  const float magT = __expf(-(float)CT * __expf(nu[h]));
  float sn, cs; __sincosf((float)CT * theta[h], &sn, &cs);
  const float tr = magT * cs, ti = magT * sn;

  float cr = h0r[h], ci = h0i[h];
  CR[(size_t)(b * NC) * HH + h] = cr;
  CI[(size_t)(b * NC) * HH + h] = ci;
  for (int c = 1; c < NC; ++c) {
    const size_t op = (size_t)(b * NC + c - 1) * HH + h;
    const float sr = SR[op], si = SI[op];
    const float nr = fmaf(tr, cr, fmaf(-ti, ci, sr));
    const float ni = fmaf(tr, ci, fmaf( ti, cr, si));
    cr = nr; ci = ni;
    const size_t oc = (size_t)(b * NC + c) * HH + h;
    CR[oc] = cr; CI[oc] = ci;
  }
}

// ---------------------------------------------------------------------------
// Kernel 2c: final per-chunk scan from carry-in; write Re(h)
// grid: (HH/256, NC, BB)
// ---------------------------------------------------------------------------
__global__ __launch_bounds__(256)
void lru_chunk_scan(const float* __restrict__ BxR, const float* __restrict__ BxI,
                    const float* __restrict__ nu, const float* __restrict__ theta,
                    const float* __restrict__ CR, const float* __restrict__ CI,
                    float* __restrict__ out) {
  const int h = blockIdx.x * blockDim.x + threadIdx.x;
  const int c = blockIdx.y;
  const int b = blockIdx.z;
  const float mag = __expf(-__expf(nu[h]));
  float sn, cs; __sincosf(theta[h], &sn, &cs);
  const float lr = mag * cs, li = mag * sn;

  const size_t oc = (size_t)(b * NC + c) * HH + h;
  float sr = CR[oc], si = CI[oc];

  const size_t base = ((size_t)b * LL + (size_t)c * CT) * HH + h;
  const float* pr = BxR + base;
  const float* pi = BxI + base;
  float* po = out + base;
#pragma unroll 4
  for (int j = 0; j < CT; ++j) {
    const float xr = pr[(size_t)j * HH];
    const float xi = pi[(size_t)j * HH];
    const float nr = fmaf(lr, sr, fmaf(-li, si, xr));
    const float ni = fmaf(lr, si, fmaf( li, sr, xi));
    sr = nr; si = ni;
    po[(size_t)j * HH] = sr;
  }
}

// ---------------------------------------------------------------------------
extern "C" void kernel_launch(void* const* d_in, const int* in_sizes, int n_in,
                              void* d_out, int out_size, void* d_ws, size_t ws_size,
                              hipStream_t stream) {
  (void)in_sizes; (void)n_in; (void)out_size; (void)ws_size;
  const float* x     = (const float*)d_in[0];
  const float* Breal = (const float*)d_in[1];
  const float* Bimg  = (const float*)d_in[2];
  const float* nu    = (const float*)d_in[3];
  const float* theta = (const float*)d_in[4];
  const float* delta = (const float*)d_in[5];
  const float* h0r   = (const float*)d_in[6];
  const float* h0i   = (const float*)d_in[7];
  float* out = (float*)d_out;

  // workspace layout
  float* BxR = (float*)d_ws;                       // MTOT*HH
  float* BxI = BxR + (size_t)MTOT * HH;            // MTOT*HH
  float* SR  = BxI + (size_t)MTOT * HH;            // BB*NC*HH
  float* SI  = SR + (size_t)BB * NC * HH;
  float* CR  = SI + (size_t)BB * NC * HH;
  float* CI  = CR + (size_t)BB * NC * HH;

  dim3 gGemm(HH / 64, MTOT / 128);
  lru_gemm_bf16<<<gGemm, 256, 0, stream>>>(x, Breal, Bimg, delta, BxR, BxI);

  dim3 gChunk(HH / 256, NC, BB);
  lru_chunk_sum<<<gChunk, 256, 0, stream>>>(BxR, BxI, nu, theta, SR, SI);

  dim3 gComb(HH / 256, BB);
  lru_chunk_combine<<<gComb, 256, 0, stream>>>(SR, SI, nu, theta, h0r, h0i, CR, CI);

  lru_chunk_scan<<<gChunk, 256, 0, stream>>>(BxR, BxI, nu, theta, CR, CI, out);
}
